// WarpNet_3204045602901
// MI455X (gfx1250) — matrix operand order repaired and checked by reading, not compiled, not run on hardware
//
#include <hip/hip_runtime.h>
#include <math.h>

// ---------------------------------------------------------------------------
// Types for CDNA5 WMMA (wave32): D(f32 16x16) = A(bf16 16x32) x B(bf16 32x16) + C
// ---------------------------------------------------------------------------
typedef __attribute__((ext_vector_type(16))) __bf16 v16bf;
typedef __attribute__((ext_vector_type(8)))  __bf16 v8bf;
typedef __attribute__((ext_vector_type(8)))  float  v8f;

__device__ __forceinline__ __bf16 f2bf(float f) {
    unsigned u = __builtin_bit_cast(unsigned, f);
    u += 0x7fffu + ((u >> 16) & 1u);          // round-to-nearest-even
    unsigned short h = (unsigned short)(u >> 16);
    return __builtin_bit_cast(__bf16, h);
}

// A fragment: 16x32 bf16, row = lane&15 of the A matrix (row-major [M][K] storage).
// ISA layout: VGPR0..3 hold K = 8*half + {0..7}; VGPR4..7 hold K = 16 + 8*half + {0..7}.
__device__ __forceinline__ v16bf load_a_frag(const __bf16* __restrict__ row, int k0, int half) {
    v8bf lo = *(const v8bf*)(row + k0 + 8 * half);
    v8bf hi = *(const v8bf*)(row + k0 + 16 + 8 * half);
    v16bf a;
#pragma unroll
    for (int i = 0; i < 8; ++i) { a[i] = lo[i]; a[8 + i] = hi[i]; }
    return a;
}

// B fragment: 32x16 bf16, col = lane&15, stored patch-major [N][K]: lane reads K = 8*half + {0..7}, 16 + 8*half + {0..7}.
__device__ __forceinline__ v16bf load_b_frag(const __bf16* __restrict__ row, int k0, int half) {
    v8bf lo = *(const v8bf*)(row + k0 + 8 * half);
    v8bf hi = *(const v8bf*)(row + k0 + 16 + 8 * half);
    v16bf b;
#pragma unroll
    for (int i = 0; i < 8; ++i) { b[i] = lo[i]; b[8 + i] = hi[i]; }
    return b;
}

__device__ __forceinline__ v8f wmma_bf16(v16bf a, v16bf b, v8f c) {
    return __builtin_amdgcn_wmma_f32_16x16x32_bf16(false, a, false, b, (short)0, c, false, false);
}

// ---------------------------------------------------------------------------
// fp32 -> bf16 weight packing (weights are already [Co][Ci*kh*kw] row-major)
// ---------------------------------------------------------------------------
__global__ void pack_bf16_kernel(const float* __restrict__ src, __bf16* __restrict__ dst, int n) {
    int i = blockIdx.x * 256 + threadIdx.x;
    if (i < n) dst[i] = f2bf(src[i]);
}

// ---------------------------------------------------------------------------
// im2col with reflect padding -> bf16, patch-major [P][K]
// ---------------------------------------------------------------------------
__global__ void im2col_kernel(const float* __restrict__ src, __bf16* __restrict__ dst,
                              int C, int H, int W, int Ho, int Wo,
                              int ks, int stride, int pad, long long total, int K) {
    long long i = (long long)blockIdx.x * 256 + threadIdx.x;
    if (i >= total) return;
    int k = (int)(i % K);
    int p = (int)(i / K);
    int ksq = ks * ks;
    int ci = k / ksq;
    int t = k - ci * ksq;
    int dy = t / ks, dx = t - dy * ks;
    int oy = p / Wo, ox = p - oy * Wo;
    int iy = oy * stride + dy - pad;
    int ix = ox * stride + dx - pad;
    iy = iy < 0 ? -iy : (iy >= H ? 2 * H - 2 - iy : iy);
    ix = ix < 0 ? -ix : (ix >= W ? 2 * W - 2 - ix : ix);
    dst[i] = f2bf(src[((size_t)ci * H + iy) * W + ix]);
}

// ---------------------------------------------------------------------------
// WMMA GEMM: C[Co][P] = A[Co][K](bf16) * B[P][K](bf16,patch-major)^T + bias
// block = 128 thr (4 waves); block tile 64(Co) x 64(P); wave tile 16 x 64; K step 32
// ---------------------------------------------------------------------------
__global__ void gemm_kernel(const __bf16* __restrict__ A, const __bf16* __restrict__ Bm,
                            const float* __restrict__ bias, float* __restrict__ C,
                            int Co, int K, int P) {
    const int wave = threadIdx.x >> 5;
    const int lane = threadIdx.x & 31;
    const int l15 = lane & 15;
    const int half = lane >> 4;
    const int co0 = blockIdx.y * 64 + wave * 16;
    const int p0 = blockIdx.x * 64;

    const __bf16* arow = A + (size_t)(co0 + l15) * K;
    const __bf16* b0 = Bm + (size_t)(p0 + l15) * K;
    const __bf16* b1 = b0 + (size_t)16 * K;
    const __bf16* b2 = b0 + (size_t)32 * K;
    const __bf16* b3 = b0 + (size_t)48 * K;

    v8f acc0 = {0.f,0.f,0.f,0.f,0.f,0.f,0.f,0.f};
    v8f acc1 = acc0, acc2 = acc0, acc3 = acc0;

    for (int k0 = 0; k0 < K; k0 += 32) {
        v16bf a = load_a_frag(arow, k0, half);
        acc0 = wmma_bf16(a, load_b_frag(b0, k0, half), acc0);
        acc1 = wmma_bf16(a, load_b_frag(b1, k0, half), acc1);
        acc2 = wmma_bf16(a, load_b_frag(b2, k0, half), acc2);
        acc3 = wmma_bf16(a, load_b_frag(b3, k0, half), acc3);
    }
#pragma unroll
    for (int r = 0; r < 8; ++r) {
        int row = co0 + r + 8 * half;
        float bi = bias ? bias[row] : 0.f;
        float* crow = C + (size_t)row * P + p0 + l15;
        crow[0]  = acc0[r] + bi;
        crow[16] = acc1[r] + bi;
        crow[32] = acc2[r] + bi;
        crow[48] = acc3[r] + bi;
    }
}

// ---------------------------------------------------------------------------
// Instance-norm stats per channel: stats[2c]=mean, stats[2c+1]=rsqrt(var+eps)
// ---------------------------------------------------------------------------
__global__ void inorm_stats_kernel(const float* __restrict__ X, float* __restrict__ stats, int P) {
    const int c = blockIdx.x;
    const float* x = X + (size_t)c * P;
    float s = 0.f, ss = 0.f;
    for (int i = threadIdx.x; i < P; i += 256) { float v = x[i]; s += v; ss += v * v; }
    __shared__ float rs[256], rq[256];
    rs[threadIdx.x] = s; rq[threadIdx.x] = ss;
    __syncthreads();
    for (int st = 128; st > 0; st >>= 1) {
        if (threadIdx.x < st) { rs[threadIdx.x] += rs[threadIdx.x + st]; rq[threadIdx.x] += rq[threadIdx.x + st]; }
        __syncthreads();
    }
    if (threadIdx.x == 0) {
        float m = rs[0] / (float)P;
        float v = rq[0] / (float)P - m * m;
        stats[2 * c] = m;
        stats[2 * c + 1] = rsqrtf(fmaxf(v, 0.f) + 1e-5f);
    }
}

// y = prelu( inorm(x) [+ residual], alpha )
__global__ void inorm_prelu_kernel(const float* __restrict__ X, const float* __restrict__ res,
                                   const float* __restrict__ stats, const float* __restrict__ alpha_p,
                                   float* __restrict__ Y, int C, int P) {
    long long i = (long long)blockIdx.x * 256 + threadIdx.x;
    if (i >= (long long)C * P) return;
    int c = (int)(i / P);
    float v = (X[i] - stats[2 * c]) * stats[2 * c + 1];
    if (res) v += res[i];
    float a = alpha_p[0];
    Y[i] = v >= 0.f ? v : a * v;
}

// nearest up x2: [C][H][W] -> [C][2H][2W]
__global__ void up2_kernel(const float* __restrict__ X, float* __restrict__ Y, int C, int H, int W) {
    int i = blockIdx.x * 256 + threadIdx.x;
    int n = C * 4 * H * W;
    if (i >= n) return;
    int W2 = 2 * W;
    int c = i / (4 * H * W);
    int rem = i - c * 4 * H * W;
    int y = rem / W2;
    int x = rem - y * W2;
    Y[i] = X[((size_t)c * H + (y >> 1)) * W + (x >> 1)];
}

// avg_pool2d(B_lab_map, 4): [2][3][256][256] -> [2][3][4096]
__global__ void avgpool4_kernel(const float* __restrict__ X, float* __restrict__ Y) {
    int i = blockIdx.x * 256 + threadIdx.x;
    if (i >= 2 * 3 * 4096) return;
    int bc = i >> 12;
    int m = i & 4095;
    int fy = m >> 6, fx = m & 63;
    const float* p = X + ((size_t)bc * 256 + fy * 4) * 256 + fx * 4;
    float s = 0.f;
#pragma unroll
    for (int a = 0; a < 4; ++a)
#pragma unroll
        for (int b = 0; b < 4; ++b) s += p[a * 256 + b];
    Y[i] = s * (1.f / 16.f);
}

// center over positions + L2 normalize over channels; emit transposed bf16 [N][C]
__global__ void embed_finalize_kernel(const float* __restrict__ T /*[256][4096]*/,
                                      const float* __restrict__ stats,
                                      __bf16* __restrict__ outT /*[4096][256]*/) {
    int n = blockIdx.x * 128 + threadIdx.x;
    if (n >= 4096) return;
    float ss = 0.f;
    for (int c = 0; c < 256; ++c) { float v = T[(size_t)c * 4096 + n] - stats[2 * c]; ss += v * v; }
    float inv = 1.f / (sqrtf(ss) + 2.220446049250313e-16f);
    for (int c = 0; c < 256; ++c) {
        float v = (T[(size_t)c * 4096 + n] - stats[2 * c]) * inv;
        outT[(size_t)n * 256 + c] = f2bf(v);
    }
}

// ---------------------------------------------------------------------------
// Fused correlation + WTA softmax + warp + up4.
// One block = 16 rows (n) of f[b]. WMMA builds f[16][4096] in 256KB LDS
// (CDNA5 WGP has 320KB), then in-LDS reductions; never spills f to HBM.
// ---------------------------------------------------------------------------
__global__ void corr_kernel(const __bf16* __restrict__ thetaT /*[2][4096][256]*/,
                            const __bf16* __restrict__ phiT   /*[2][4096][256]*/,
                            const float* __restrict__ blab    /*[2][3][4096]*/,
                            float* __restrict__ outY          /*[2][3][256][256]*/,
                            float* __restrict__ outS          /*[2][1][256][256]*/) {
    const int b = blockIdx.z;
    const int n0 = blockIdx.x * 16;
    const int wave = threadIdx.x >> 5;
    const int lane = threadIdx.x & 31;
    const int l15 = lane & 15;
    const int half = lane >> 4;

    extern __shared__ float fls[];  // [16][4096] = 256 KB

    // Preload all 8 K-chunks of the A fragment (theta rows n0..n0+15)
    const __bf16* arow = thetaT + ((size_t)b * 4096 + n0 + l15) * 256;
    v16bf afr[8];
#pragma unroll
    for (int kk = 0; kk < 8; ++kk) afr[kk] = load_a_frag(arow, kk * 32, half);

    const __bf16* pB = phiT + (size_t)b * 4096 * 256;
    for (int mt = wave; mt < 256; mt += 4) {
        const __bf16* brow = pB + (size_t)(mt * 16 + l15) * 256;
        v8f acc = {0.f,0.f,0.f,0.f,0.f,0.f,0.f,0.f};
#pragma unroll
        for (int kk = 0; kk < 8; ++kk)
            acc = wmma_bf16(afr[kk], load_b_frag(brow, kk * 32, half), acc);
#pragma unroll
        for (int r = 0; r < 8; ++r)
            fls[(size_t)(r + 8 * half) * 4096 + mt * 16 + l15] = acc[r];
    }
    __syncthreads();

    __shared__ float red[128];
    __shared__ float s_max[16], s_mw[16], s_sum[16], s_y0[16], s_y1[16], s_y2[16];

    const int r = threadIdx.x >> 3;   // row 0..15
    const int sub = threadIdx.x & 7;  // 8 workers per row
    const float* frow = fls + (size_t)r * 4096;

    // pass 1: row max of f
    float pm = -3.0e38f;
    for (int m = sub; m < 4096; m += 8) pm = fmaxf(pm, frow[m]);
    red[threadIdx.x] = pm; __syncthreads();
    if (sub == 0) { float v = red[r * 8]; for (int i = 1; i < 8; ++i) v = fmaxf(v, red[r * 8 + i]); s_max[r] = v; }
    __syncthreads();
    const float rmax = s_max[r];

    // pass 2: max of WTA-scaled values
    pm = -3.0e38f;
    for (int m = sub; m < 4096; m += 8) {
        float f = frow[m];
        float w = (f == rmax) ? f : f * 1e-4f;
        pm = fmaxf(pm, w);
    }
    red[threadIdx.x] = pm; __syncthreads();
    if (sub == 0) { float v = red[r * 8]; for (int i = 1; i < 8; ++i) v = fmaxf(v, red[r * 8 + i]); s_mw[r] = v; }
    __syncthreads();
    const float mw = s_mw[r];

    // pass 3: exp-sum and warp accumulation against blab
    const float* bl = blab + (size_t)b * 3 * 4096;
    float ps = 0.f, py0 = 0.f, py1 = 0.f, py2 = 0.f;
    for (int m = sub; m < 4096; m += 8) {
        float f = frow[m];
        float w = (f == rmax) ? f : f * 1e-4f;
        float e = __expf((w - mw) * 200.f);  // 1/TEMP, TEMP = 0.005
        ps += e;
        py0 += e * bl[m];
        py1 += e * bl[4096 + m];
        py2 += e * bl[8192 + m];
    }
    red[threadIdx.x] = ps; __syncthreads();
    if (sub == 0) { float v = 0.f; for (int i = 0; i < 8; ++i) v += red[r * 8 + i]; s_sum[r] = v; } __syncthreads();
    red[threadIdx.x] = py0; __syncthreads();
    if (sub == 0) { float v = 0.f; for (int i = 0; i < 8; ++i) v += red[r * 8 + i]; s_y0[r] = v; } __syncthreads();
    red[threadIdx.x] = py1; __syncthreads();
    if (sub == 0) { float v = 0.f; for (int i = 0; i < 8; ++i) v += red[r * 8 + i]; s_y1[r] = v; } __syncthreads();
    red[threadIdx.x] = py2; __syncthreads();
    if (sub == 0) { float v = 0.f; for (int i = 0; i < 8; ++i) v += red[r * 8 + i]; s_y2[r] = v; } __syncthreads();

    // pass 4: emit up4-replicated outputs
    if (threadIdx.x < 16) {
        int rr = threadIdx.x;
        int n = n0 + rr;
        int fy = n >> 6, fx = n & 63;
        float inv = 1.f / s_sum[rr];
        float yv[3] = { s_y0[rr] * inv, s_y1[rr] * inv, s_y2[rr] * inv };
        for (int c = 0; c < 3; ++c) {
            float* op = outY + (((size_t)b * 3 + c) * 256 + 4 * fy) * 256 + 4 * fx;
            for (int i = 0; i < 4; ++i)
                for (int j = 0; j < 4; ++j) op[i * 256 + j] = yv[c];
        }
        float sv = s_max[rr];
        float* sp = outS + ((size_t)b * 256 + 4 * fy) * 256 + 4 * fx;
        for (int i = 0; i < 4; ++i)
            for (int j = 0; j < 4; ++j) sp[i * 256 + j] = sv;
    }
}

// ---------------------------------------------------------------------------
// Host orchestration
// ---------------------------------------------------------------------------
extern "C" void kernel_launch(void* const* d_in, const int* in_sizes, int n_in,
                              void* d_out, int out_size, void* d_ws, size_t ws_size,
                              hipStream_t stream) {
    (void)in_sizes; (void)n_in; (void)out_size; (void)ws_size;

    const float* B_lab = (const float*)d_in[0];
    const float* relu[2][4] = {
        { (const float*)d_in[1], (const float*)d_in[2], (const float*)d_in[3], (const float*)d_in[4] },  // A tower
        { (const float*)d_in[5], (const float*)d_in[6], (const float*)d_in[7], (const float*)d_in[8] },  // B tower
    };

    // params, recursive dict-insertion order
    // l2:9..14  l3:15..20  l4:21..26  l5:27..32  res0:33..37 res1:38..42 res2:43..47 theta:48,49 phi:50,51
    const float* bia[8];   // layer conv biases: l2b1,l2b2,l3b1,l3b2,l4b1,l4b2,l5b1,l5b2
    const float* alp[8];   // layer prelu alphas
    bia[0]=(const float*)d_in[10]; alp[0]=(const float*)d_in[11];
    bia[1]=(const float*)d_in[13]; alp[1]=(const float*)d_in[14];
    bia[2]=(const float*)d_in[16]; alp[2]=(const float*)d_in[17];
    bia[3]=(const float*)d_in[19]; alp[3]=(const float*)d_in[20];
    bia[4]=(const float*)d_in[22]; alp[4]=(const float*)d_in[23];
    bia[5]=(const float*)d_in[25]; alp[5]=(const float*)d_in[26];
    bia[6]=(const float*)d_in[28]; alp[6]=(const float*)d_in[29];
    bia[7]=(const float*)d_in[31]; alp[7]=(const float*)d_in[32];
    const float* res_b1[3] = { (const float*)d_in[34], (const float*)d_in[39], (const float*)d_in[44] };
    const float* res_b2[3] = { (const float*)d_in[36], (const float*)d_in[41], (const float*)d_in[46] };
    const float* res_a[3]  = { (const float*)d_in[37], (const float*)d_in[42], (const float*)d_in[47] };
    const float* theta_b = (const float*)d_in[49];
    const float* phi_b   = (const float*)d_in[51];

    // ---- workspace carve-up ----
    char* ws = (char*)d_ws;
    size_t off = 0;
    auto alloc = [&](size_t bytes) -> char* {
        off = (off + 255) & ~(size_t)255;
        char* p = ws + off;
        off += bytes;
        return p;
    };

    // packed bf16 weights
    struct WP { int idx; int n; };
    const WP wps[16] = {
        {9,147456},{12,73728},{15,294912},{18,73728},{21,1179648},{24,147456},
        {27,1179648},{30,147456},{33,589824},{35,589824},{38,589824},{40,589824},
        {43,589824},{45,589824},{48,65536},{50,65536} };
    __bf16* wpk[16];
    for (int i = 0; i < 16; ++i) {
        wpk[i] = (__bf16*)alloc((size_t)wps[i].n * 2);
        pack_bf16_kernel<<<(wps[i].n + 255) / 256, 256, 0, stream>>>(
            (const float*)d_in[wps[i].idx], wpk[i], wps[i].n);
    }
    const __bf16* wl[8]     = { wpk[0], wpk[1], wpk[2], wpk[3], wpk[4], wpk[5], wpk[6], wpk[7] };
    const __bf16* w_res1[3] = { wpk[8], wpk[10], wpk[12] };
    const __bf16* w_res2[3] = { wpk[9], wpk[11], wpk[13] };
    const __bf16* w_theta = wpk[14];
    const __bf16* w_phi   = wpk[15];

    __bf16* colbuf = (__bf16*)alloc((size_t)16384 * 1152 * 2);     // 37.75 MB, reused
    float* tmp1 = (float*)alloc((size_t)2097152 * 4);              // 8 MB
    float* tmp2 = (float*)alloc((size_t)2097152 * 4);              // 8 MB
    float* feat[2][2];                                             // [tower][pingpong], each [2][256][4096]
    for (int t = 0; t < 2; ++t)
        for (int p = 0; p < 2; ++p) feat[t][p] = (float*)alloc((size_t)2 * 256 * 4096 * 4);
    __bf16* thetaT = (__bf16*)alloc((size_t)2 * 4096 * 256 * 2);
    __bf16* phiT   = (__bf16*)alloc((size_t)2 * 4096 * 256 * 2);
    float* statbuf = (float*)alloc(512 * 4);
    float* blab    = (float*)alloc((size_t)2 * 3 * 4096 * 4);

    // ---- helpers ----
    auto conv = [&](const float* src, int C, int H, int W, int ks, int stride,
                    const __bf16* wp, const float* bias, int Co, float* out) {
        int pad = (ks == 3) ? 1 : 0;
        int Ho = (H + 2 * pad - ks) / stride + 1;
        int Wo = (W + 2 * pad - ks) / stride + 1;
        int K = C * ks * ks;
        int P = Ho * Wo;
        long long total = (long long)P * K;
        im2col_kernel<<<(unsigned)((total + 255) / 256), 256, 0, stream>>>(
            src, colbuf, C, H, W, Ho, Wo, ks, stride, pad, total, K);
        gemm_kernel<<<dim3(P / 64, Co / 64), 128, 0, stream>>>(wp, colbuf, bias, out, Co, K, P);
    };
    auto inp = [&](const float* x, int C, int P, const float* alpha, const float* res, float* dst) {
        inorm_stats_kernel<<<C, 256, 0, stream>>>(x, statbuf, P);
        long long n = (long long)C * P;
        inorm_prelu_kernel<<<(unsigned)((n + 255) / 256), 256, 0, stream>>>(x, res, statbuf, alpha, dst, C, P);
    };
    auto up2 = [&](const float* x, int C, int H, int W, float* dst) {
        int n = C * 4 * H * W;
        up2_kernel<<<(n + 255) / 256, 256, 0, stream>>>(x, dst, C, H, W);
    };

    // ---- feature towers ----
    float* towerOut[2];
    for (int t = 0; t < 2; ++t) {
        float* f0 = feat[t][0];
        float* f1 = feat[t][1];
        for (int b = 0; b < 2; ++b) {
            float* fd = f0 + (size_t)b * 256 * 4096;
            // layer2: 128x128x128 -> conv128 -> conv64 s2 -> 64ch @64x64
            conv(relu[t][0] + (size_t)b * 128 * 128 * 128, 128, 128, 128, 3, 1, wl[0], bia[0], 128, tmp1);
            inp(tmp1, 128, 16384, alp[0], nullptr, tmp1);
            conv(tmp1, 128, 128, 128, 3, 2, wl[1], bia[1], 64, tmp2);
            inp(tmp2, 64, 4096, alp[1], nullptr, fd);
            // layer3: 256 @64x64 -> 128 -> 64
            conv(relu[t][1] + (size_t)b * 256 * 4096, 256, 64, 64, 3, 1, wl[2], bia[2], 128, tmp1);
            inp(tmp1, 128, 4096, alp[2], nullptr, tmp1);
            conv(tmp1, 128, 64, 64, 3, 1, wl[3], bia[3], 64, tmp2);
            inp(tmp2, 64, 4096, alp[3], nullptr, fd + (size_t)64 * 4096);
            // layer4: 512 @32x32 -> 256 -> 64 -> up2
            conv(relu[t][2] + (size_t)b * 512 * 1024, 512, 32, 32, 3, 1, wl[4], bia[4], 256, tmp1);
            inp(tmp1, 256, 1024, alp[4], nullptr, tmp1);
            conv(tmp1, 256, 32, 32, 3, 1, wl[5], bia[5], 64, tmp2);
            inp(tmp2, 64, 1024, alp[5], nullptr, tmp2);
            up2(tmp2, 64, 32, 32, fd + (size_t)128 * 4096);
            // layer5: 512 @16x16 -> 256 -> up2 -> 64 -> up2
            conv(relu[t][3] + (size_t)b * 512 * 256, 512, 16, 16, 3, 1, wl[6], bia[6], 256, tmp1);
            inp(tmp1, 256, 256, alp[6], nullptr, tmp1);
            up2(tmp1, 256, 16, 16, tmp2);
            conv(tmp2, 256, 32, 32, 3, 1, wl[7], bia[7], 64, tmp1);
            inp(tmp1, 64, 1024, alp[7], nullptr, tmp1);
            up2(tmp1, 64, 32, 32, fd + (size_t)192 * 4096);
        }
        // resblocks
        float* cur = f0;
        float* nxt = f1;
        for (int rb = 0; rb < 3; ++rb) {
            for (int b = 0; b < 2; ++b) {
                float* xb = cur + (size_t)b * 256 * 4096;
                float* ob = nxt + (size_t)b * 256 * 4096;
                conv(xb, 256, 64, 64, 3, 1, w_res1[rb], res_b1[rb], 256, tmp1);
                inp(tmp1, 256, 4096, res_a[rb], nullptr, tmp1);
                conv(tmp1, 256, 64, 64, 3, 1, w_res2[rb], res_b2[rb], 256, tmp2);
                inp(tmp2, 256, 4096, res_a[rb], xb, ob);   // inorm + residual + prelu
            }
            float* s = cur; cur = nxt; nxt = s;
        }
        towerOut[t] = cur;
    }

    // ---- embeddings (1x1 conv + center + L2 normalize), transposed bf16 output ----
    auto embed = [&](const float* featp, const __bf16* wp, const float* bias, __bf16* outT) {
        for (int b = 0; b < 2; ++b) {
            conv(featp + (size_t)b * 256 * 4096, 256, 64, 64, 1, 1, wp, bias, 256, tmp1);
            inorm_stats_kernel<<<256, 256, 0, stream>>>(tmp1, statbuf, 4096);  // only mean used
            embed_finalize_kernel<<<32, 128, 0, stream>>>(tmp1, statbuf, outT + (size_t)b * 4096 * 256);
        }
    };
    embed(towerOut[0], w_theta, theta_b, thetaT);
    embed(towerOut[1], w_phi, phi_b, phiT);

    // ---- avg-pool Lab map ----
    avgpool4_kernel<<<(2 * 3 * 4096 + 255) / 256, 256, 0, stream>>>(B_lab, blab);

    // ---- fused correlation + WTA softmax + warp + up4 ----
    float* outY = (float*)d_out;                 // [2][3][256][256]
    float* outS = outY + (size_t)2 * 3 * 256 * 256;  // [2][1][256][256]
    corr_kernel<<<dim3(256, 1, 2), 128, 16 * 4096 * sizeof(float), stream>>>(
        thetaT, phiT, blab, outY, outS);
}
